// Att_patch_net_83270825935643
// MI455X (gfx1250) — matrix operand order repaired and checked by reading, not compiled, not run on hardware
//
#include <hip/hip_runtime.h>
#include <hip/hip_bf16.h>
#include <math.h>

// ---------------- problem constants ----------------
#define BB    32
#define CIN   2048
#define HH    14
#define WW    14
#define HW    196          // 14*14
#define DCH   128
#define KC    32           // codewords
#define NCLS  1000
#define WIN   5
#define PHW   10           // ph == pw
#define NPIX  25           // WIN*WIN
#define KD    4096         // K*D
#define BN_EPS 1e-5f

typedef __attribute__((ext_vector_type(16))) __bf16 v16bf;
typedef __attribute__((ext_vector_type(8)))  float  v8f;
typedef __attribute__((ext_vector_type(2)))  float  v2f;

union BF16x16 { unsigned u[8]; v16bf v; };

__device__ __forceinline__ unsigned pack_bf16(float lo, float hi) {
  unsigned a = __builtin_bit_cast(unsigned, lo) >> 16;
  unsigned b = __builtin_bit_cast(unsigned, hi) & 0xFFFF0000u;
  return a | b;
}

// ---- CDNA5 async global->LDS copy (ASYNCcnt-tracked), ISA ch.10/15.18 ----
__device__ __forceinline__ void async_ld_b128(void* lds_dst, const void* gsrc) {
  unsigned loff = (unsigned)(size_t)lds_dst;   // generic LDS addr: low 32 bits
  asm volatile("global_load_async_to_lds_b128 %0, %1, off"
               :: "v"(loff), "v"(gsrc) : "memory");
}
__device__ __forceinline__ void wait_async0() {
  asm volatile("s_wait_asynccnt 0x0" ::: "memory");
}

// =====================================================================
// Kernel 1: 1x1 conv as GEMM (bf16 WMMA, f32 accumulate) + BN + ReLU.
// y[b,hw,d] = relu(bn(sum_c w[d,c] * x[b,c,hw] + cb[d])), stored d-contiguous.
// Block: 256 thr = 8 waves; wave w owns D rows [16w,16w+16); block covers a
// 16-wide hw tile for one batch. K loop: 2048 in steps of 32.
// Out-of-range hw columns: address clamped (garbage only pollutes the WMMA
// output column that is never stored) -> no divergent branches around loads.
// =====================================================================
__global__ __launch_bounds__(256) void conv_bn_relu_wmma(
    const float* __restrict__ x, const float* __restrict__ w,
    const float* __restrict__ cb, const float* __restrict__ gamma,
    const float* __restrict__ beta, const float* __restrict__ mean,
    const float* __restrict__ var, float* __restrict__ yT)
{
  const int b   = blockIdx.y;
  const int nt  = blockIdx.x;            // hw tile (0..12)
  const int wv  = threadIdx.x >> 5;      // wave -> D tile
  const int L   = threadIdx.x & 31;
  const int lh  = L >> 4;                // lane half
  const int ln  = L & 15;

  const int hw0 = nt * 16;
  const int d0  = wv * 16;
  const float* xb = x + (size_t)b * CIN * HW;

  const int hwc  = hw0 + ln;             // B/C column for this lane
  const bool hok = hwc < HW;
  const int hwcc = hok ? hwc : (HW - 1); // clamped load address
  const int am    = d0 + ln;             // A row (output channel)
  const int khalf = lh * 8;

  v8f acc = {};
  for (int k0 = 0; k0 < CIN; k0 += 32) {
    // prefetch next K-slab of the streamed activation tile
    if (k0 + 32 < CIN)
      __builtin_prefetch(xb + (size_t)(k0 + 32) * HW + hw0, 0, 1);

    BF16x16 A, Bf;
    // A: w[am, k0+koff] ; ISA 16-bit A layout (K 0-7/16-23 vs 8-15/24-31)
    #pragma unroll
    for (int q = 0; q < 8; ++q) {
      int koff = (q < 4) ? (khalf + 2 * q) : (16 + khalf + 2 * (q - 4));
      const float* p = w + (size_t)am * CIN + k0 + koff;
      A.u[q] = pack_bf16(p[0], p[1]);
    }
    // B: x[b, k0+kb, hwcc] ; column per lane, K halves across lane halves
    #pragma unroll
    for (int q = 0; q < 8; ++q) {
      int kb = (q < 4) ? (lh * 8 + 2 * q) : (16 + lh * 8 + 2 * (q - 4));
      float lo = xb[(size_t)(k0 + kb)     * HW + hwcc];
      float hi = xb[(size_t)(k0 + kb + 1) * HW + hwcc];
      Bf.u[q] = pack_bf16(lo, hi);
    }
    acc = __builtin_amdgcn_wmma_f32_16x16x32_bf16(
        false, A.v, false, Bf.v, (short)0, acc, false, false);
  }

  if (hok) {
    // 8 C rows per lane are consecutive d: two 128-bit stores
    float r[8];
    #pragma unroll
    for (int q = 0; q < 8; ++q) {
      int d = d0 + q + 8 * lh;
      float val = acc[q] + cb[d];
      float inv = gamma[d] * rsqrtf(var[d] + BN_EPS);
      val = (val - mean[d]) * inv + beta[d];
      r[q] = fmaxf(val, 0.f);
    }
    float* outp = yT + ((size_t)b * HW + hwc) * DCH + d0 + 8 * lh;
    *(float4*)(outp)     = make_float4(r[0], r[1], r[2], r[3]);
    *(float4*)(outp + 4) = make_float4(r[4], r[5], r[6], r[7]);
  }
}

// =====================================================================
// Kernel 2: zero the v accumulator
// =====================================================================
__global__ void zero_kernel(float* __restrict__ p, int n) {
  int i = blockIdx.x * blockDim.x + threadIdx.x;
  if (i < n) p[i] = 0.f;
}

// =====================================================================
// Kernel 3: soft-VQ encoding per patch, f32 WMMA (16x16x4), LDS-resident.
// One 128-thread block (4 waves) handles 10 patches of one batch image.
// X and codewords are staged with GLOBAL_LOAD_ASYNC_TO_LDS_B128.
// =====================================================================
__global__ __launch_bounds__(128) void encode_wmma(
    const float* __restrict__ yT, const float* __restrict__ codew,
    const float* __restrict__ scale, float* __restrict__ vout)
{
  __shared__ float Xs[32][DCH];      // patch pixels (rows >=25 stay zero)
  __shared__ float Cs[KC][DCH];      // codewords
  __shared__ float Al[32][KC];       // logits -> softmax assignments
  __shared__ float x2s[32], c2s[KC], asum[KC], sc[KC];
  __shared__ float vacc[KC * DCH];   // per-image partial sum over patches
  __shared__ float enorm2, rnorm;

  const int b  = blockIdx.y;
  const int pg = blockIdx.x;         // patch group (10 patches each)
  const int t  = threadIdx.x;
  const int wv = t >> 5, L = t & 31, lh = L >> 4, ln = L & 15;

  // async-stage codewords into LDS; zero vacc + X pad rows meanwhile
  for (int i = t * 4; i < KC * DCH; i += 512)
    async_ld_b128(&((float*)Cs)[i], codew + i);
  for (int i = t; i < KC * DCH; i += 128) vacc[i] = 0.f;
  for (int i = t; i < 7 * DCH; i += 128) Xs[25 + (i >> 7)][i & 127] = 0.f;
  if (t == 0) enorm2 = 0.f;
  wait_async0();
  __syncthreads();
  if (t < KC) {
    float s = 0.f;
    for (int d = 0; d < DCH; ++d) s += Cs[t][d] * Cs[t][d];
    c2s[t] = s; sc[t] = scale[t];
  }
  __syncthreads();

  for (int pp = 0; pp < PHW; ++pp) {
    const int p  = pg * PHW + pp;
    const int pi = p / PHW, pj = p % PHW;

    // ---- async-load X (25 rows x 128 f32 = 800 float4 chunks) ----
    for (int j = t; j < NPIX * 32; j += 128) {
      int row = j >> 5, c4 = (j & 31) * 4;
      int pix = (pi + row / WIN) * WW + (pj + row % WIN);
      async_ld_b128(&Xs[row][c4], yT + ((size_t)b * HW + pix) * DCH + c4);
    }
    wait_async0();
    __syncthreads();
    if (t < 32) {
      float s = 0.f;
      for (int d = 0; d < DCH; ++d) s += Xs[t][d] * Xs[t][d];
      x2s[t] = s;
    }
    __syncthreads();

    // ---- xc = X * C^T  (M=32 rows of X, N=32 codewords, K=128), f32 WMMA
    {
      int mt = wv >> 1, ntk = wv & 1;      // wave -> one 16x16 tile
      int r  = mt * 16 + ln;               // A row (pixel)
      int kc = ntk * 16 + ln;              // B column (codeword)
      v8f acc = {};
      for (int kk = 0; kk < DCH; kk += 4) {
        v2f a, bb;
        a.x  = Xs[r][kk + 2 * lh];
        a.y  = Xs[r][kk + 2 * lh + 1];
        bb.x = Cs[kc][kk + 2 * lh];
        bb.y = Cs[kc][kk + 2 * lh + 1];
        acc = __builtin_amdgcn_wmma_f32_16x16x4_f32(
            false, a, false, bb, (short)0, acc, false, false);
      }
      #pragma unroll
      for (int q = 0; q < 8; ++q) {
        int rr  = mt * 16 + q + 8 * lh;
        int kcc = ntk * 16 + ln;
        float dist = x2s[rr] - 2.f * acc[q] + c2s[kcc];
        Al[rr][kcc] = sc[kcc] * dist;      // logits
      }
    }
    __syncthreads();

    // ---- softmax over K per pixel row; zero padded rows ----
    if (t < 32) {
      if (t < NPIX) {
        float m = -1e30f;
        for (int k = 0; k < KC; ++k) m = fmaxf(m, Al[t][k]);
        float s = 0.f;
        for (int k = 0; k < KC; ++k) { float e = __expf(Al[t][k] - m); Al[t][k] = e; s += e; }
        float inv = 1.f / s;
        for (int k = 0; k < KC; ++k) Al[t][k] *= inv;
      } else {
        for (int k = 0; k < KC; ++k) Al[t][k] = 0.f;
      }
    }
    __syncthreads();
    if (t < KC) {
      float s = 0.f;
      for (int n = 0; n < NPIX; ++n) s += Al[n][t];
      asum[t] = s;
    }
    __syncthreads();

    // ---- E = A^T * X - asum*C  (M=32 codewords, N=128 dims, K=25->28) ----
    float ev[4][8];
    const int kt = wv & 1;                 // codeword tile
    const int dg = wv >> 1;                // dim-tile group (4 tiles)
    float sq = 0.f;
    for (int dt4 = 0; dt4 < 4; ++dt4) {
      int dcol = (dg * 4 + dt4) * 16 + ln;
      v8f acc = {};
      for (int nn = 0; nn < 28; nn += 4) { // rows 25..27 are zero-padded
        v2f a, bb;
        int kr = kt * 16 + ln;
        a.x  = Al[nn + 2 * lh][kr];        // A^T[k, n]
        a.y  = Al[nn + 2 * lh + 1][kr];
        bb.x = Xs[nn + 2 * lh][dcol];
        bb.y = Xs[nn + 2 * lh + 1][dcol];
        acc = __builtin_amdgcn_wmma_f32_16x16x4_f32(
            false, a, false, bb, (short)0, acc, false, false);
      }
      #pragma unroll
      for (int q = 0; q < 8; ++q) {
        int kr = kt * 16 + q + 8 * lh;
        float e = acc[q] - asum[kr] * Cs[kr][dcol];
        ev[dt4][q] = e;
        sq += e * e;
      }
    }
    atomicAdd(&enorm2, sq);
    __syncthreads();
    if (t == 0) {
      rnorm = 1.f / fmaxf(sqrtf(enorm2), 1e-12f);
      enorm2 = 0.f;
    }
    __syncthreads();
    {
      float rn = rnorm;                    // each wave owns disjoint (k,d) region
      for (int dt4 = 0; dt4 < 4; ++dt4) {
        int dcol = (dg * 4 + dt4) * 16 + ln;
        #pragma unroll
        for (int q = 0; q < 8; ++q) {
          int kr = kt * 16 + q + 8 * lh;
          vacc[kr * DCH + dcol] += ev[dt4][q] * rn;
        }
      }
    }
    __syncthreads();
  }

  for (int i = t; i < KC * DCH; i += 128)
    atomicAdd(&vout[(size_t)b * KD + i], vacc[i]);
}

// =====================================================================
// Kernel 4: classifier GEMM out[32,1000] = v[32,4096] @ cls_w^T + b
// bf16 WMMA 16x16x32; 128 thr = 4 waves = 2x2 tiles per 32-col block.
// Out-of-range class columns: clamped load address, store guarded.
// =====================================================================
__global__ __launch_bounds__(128) void cls_wmma(
    const float* __restrict__ v, const float* __restrict__ W,
    const float* __restrict__ bias, float* __restrict__ out)
{
  const int nb = blockIdx.x;               // 32 blocks of 32 class columns
  const int wv = threadIdx.x >> 5, L = threadIdx.x & 31;
  const int lh = L >> 4, ln = L & 15;
  const int mt = wv >> 1, nt2 = wv & 1;

  const int col  = nb * 32 + nt2 * 16 + ln; // class column (B/C col)
  const bool cok = col < NCLS;
  const int colc = cok ? col : (NCLS - 1);  // clamped load address
  const int row  = mt * 16 + ln;            // batch row (A row)
  const int khalf = lh * 8;
  const float* Wc = W + (size_t)colc * KD;

  v8f acc = {};
  for (int k0 = 0; k0 < KD; k0 += 32) {
    if (k0 + 32 < KD)
      __builtin_prefetch(Wc + k0 + 32, 0, 1);
    BF16x16 A, Bf;
    #pragma unroll
    for (int q = 0; q < 8; ++q) {
      int koff = (q < 4) ? (khalf + 2 * q) : (16 + khalf + 2 * (q - 4));
      const float* p = v + (size_t)row * KD + k0 + koff;
      A.u[q] = pack_bf16(p[0], p[1]);
    }
    #pragma unroll
    for (int q = 0; q < 8; ++q) {
      int kb = (q < 4) ? (lh * 8 + 2 * q) : (16 + lh * 8 + 2 * (q - 4));
      Bf.u[q] = pack_bf16(Wc[k0 + kb], Wc[k0 + kb + 1]);
    }
    acc = __builtin_amdgcn_wmma_f32_16x16x32_bf16(
        false, A.v, false, Bf.v, (short)0, acc, false, false);
  }
  if (cok) {
    #pragma unroll
    for (int q = 0; q < 8; ++q) {
      int bidx = mt * 16 + q + 8 * lh;
      out[(size_t)bidx * NCLS + col] = acc[q] + bias[col];
    }
  }
}

// =====================================================================
extern "C" void kernel_launch(void* const* d_in, const int* in_sizes, int n_in,
                              void* d_out, int out_size, void* d_ws, size_t ws_size,
                              hipStream_t stream) {
  const float* x      = (const float*)d_in[0];
  const float* conv_w = (const float*)d_in[1];
  const float* conv_b = (const float*)d_in[2];
  const float* gamma  = (const float*)d_in[3];
  const float* beta   = (const float*)d_in[4];
  const float* mean   = (const float*)d_in[5];
  const float* var    = (const float*)d_in[6];
  const float* codew  = (const float*)d_in[7];
  const float* scale  = (const float*)d_in[8];
  const float* cls_w  = (const float*)d_in[9];
  const float* cls_b  = (const float*)d_in[10];
  float* out = (float*)d_out;

  float* yT   = (float*)d_ws;                         // [32][196][128] f32
  float* vbuf = yT + (size_t)BB * HW * DCH;           // [32][4096]     f32

  conv_bn_relu_wmma<<<dim3((HW + 15) / 16, BB), 256, 0, stream>>>(
      x, conv_w, conv_b, gamma, beta, mean, var, yT);

  int nv = BB * KD;
  zero_kernel<<<(nv + 255) / 256, 256, 0, stream>>>(vbuf, nv);

  encode_wmma<<<dim3(PHW, BB), 128, 0, stream>>>(yT, codew, scale, vbuf);

  cls_wmma<<<32, 128, 0, stream>>>(vbuf, cls_w, cls_b, out);
}
